// LS2T_54365696032796
// MI455X (gfx1250) — compile-verified, matrix operand order checked by reading, not run on hardware
//
#include <hip/hip_runtime.h>

typedef unsigned short u16;
typedef __attribute__((ext_vector_type(16))) __bf16 v16bf;
typedef __attribute__((ext_vector_type(16))) u16    v16u;
typedef __attribute__((ext_vector_type(8)))  float  v8f;
typedef int b128_t __attribute__((vector_size(16)));   // matches async-LDS builtin param

#define B_SZ   32
#define T_SZ   2048
#define F_SZ   512
#define U_SZ   128
#define K_T    10
#define CHUNKS 16
#define TC     128          // timesteps per chunk
#define ST     32           // timesteps per super-tile (2 WMMA row-tiles)
#define NST    (TC / ST)    // 4 super-tiles per chunk
#define APITCH 520          // u16 pitch of staged-X row (512 + pad)
#define UPITCH 132          // float pitch of M tile rows (128 + pad)
#define NSLOT  20
#define PACKED_ELEMS (K_T * 16 * 8 * 32 * 16)   // 655360 bf16
#define PACKED_BYTES (PACKED_ELEMS * 2)          // 1310720 (256B aligned)
#define MBUF_BYTES (K_T * ST * UPITCH * 4)       // 168960

#if defined(__gfx1250__) && \
    __has_builtin(__builtin_amdgcn_global_load_async_to_lds_b128) && \
    __has_builtin(__builtin_amdgcn_s_wait_asynccnt)
#define HAVE_ASYNC_LDS 1
#else
#define HAVE_ASYNC_LDS 0
#endif

__device__ __forceinline__ u16 f2bf(float f) {
    unsigned int u = __float_as_uint(f);
    u = u + 0x7FFFu + ((u >> 16) & 1u);   // round-to-nearest-even
    return (u16)(u >> 16);
}

// ---------------------------------------------------------------------------
// Pre-pack kernel weights fp32 -> bf16 in WMMA B-fragment order:
// packed[(((k*16+ft)*8+nt)*32+lane)*16 + h]
// ---------------------------------------------------------------------------
__global__ __launch_bounds__(256)
void pack_weights(const float* __restrict__ kern, u16* __restrict__ packedB) {
    int idx = blockIdx.x * 256 + threadIdx.x;
    if (idx >= PACKED_ELEMS) return;
    int h    = idx & 15;
    int lane = (idx >> 4) & 31;
    int nt   = (idx >> 9) & 7;
    int ft   = (idx >> 12) & 15;
    int k    = idx >> 16;
    int base = ft * 32 + ((lane >> 4) ? 8 : 0);
    int kk   = (h < 8) ? (base + h) : (base + 8 + h);
    int u    = nt * 16 + (lane & 15);
    packedB[idx] = f2bf(kern[((size_t)k * F_SZ + kk) * U_SZ + u]);
}

// ---------------------------------------------------------------------------
// Main fused kernel: per (batch, T-chunk): bf16 WMMA lift + in-chunk
// sub-chain scan partials. 8 waves; each owns 10 (k, u-tile) columns and
// 32 rows per B-fragment load (2 WMMAs per fragment -> halves L2 traffic).
// ---------------------------------------------------------------------------
__global__ __launch_bounds__(256)
void ls2t_main(const float* __restrict__ X, const float* __restrict__ bias,
               const u16* __restrict__ packedB, float* __restrict__ partials) {
    // Mbuf and the raw-X async staging buffer are aliased (disjoint lifetimes)
    __shared__ alignas(16) char smem[MBUF_BYTES];
    __shared__ alignas(16) u16  Abuf[ST * APITCH];
    float* Mbuf = (float*)smem;
#if HAVE_ASYNC_LDS
    float* Xraw = (float*)smem;     // ST*F_SZ*4 = 64KB <= MBUF_BYTES
#endif

    const int tid  = threadIdx.x;
    const int lane = tid & 31;
    const int wave = tid >> 5;                     // 8 waves
    const int b    = blockIdx.x / CHUNKS;
    const int c    = blockIdx.x % CHUNKS;
    const int t0   = c * TC;

    const int m  = lane & 15;
    const int hi = lane >> 4;

    // per-thread scan state: grp0 = levels 1&4, grp1 = levels 2&3 (u = tid&127)
    const int grp = tid >> 7;
    const int u   = tid & 127;
    float y1 = 0.f;
    float s11=0,s12=0,s13=0,s14=0,s22=0,s23=0,s24=0,s33=0,s34=0,s44=0; // level 4
    float t11=0,t12=0,t22=0;                                           // level 2
    float q11=0,q12=0,q13=0,q22=0,q23=0,q33=0;                         // level 3

    v8f acc0[10], acc1[10];

    for (int st = 0; st < NST; ++st) {
        const float* xbase = X + ((size_t)b * T_SZ + t0 + st * ST) * F_SZ;

        // ---- stage 32 x 512 fp32 X rows into LDS as bf16 ----
#if HAVE_ASYNC_LDS
        #pragma unroll
        for (int i = 0; i < 16; ++i) {
            int idx = tid + i * 256;          // 4096 x b128
            int row = idx >> 7;
            int c4  = idx & 127;
            void* gp = (void*)(xbase + (size_t)row * F_SZ + c4 * 4);
            float* lp = Xraw + (size_t)row * F_SZ + c4 * 4;
            __builtin_amdgcn_global_load_async_to_lds_b128(
                (__attribute__((address_space(1))) b128_t*)gp,
                (__attribute__((address_space(3))) b128_t*)(void*)lp,
                0, 0);
        }
        __builtin_amdgcn_s_wait_asynccnt(0);
        __syncthreads();
        #pragma unroll
        for (int i = 0; i < 16; ++i) {
            int idx = tid + i * 256;
            int row = idx >> 7;
            int c4  = idx & 127;
            const float4 v =
                reinterpret_cast<const float4*>(Xraw + (size_t)row * F_SZ)[c4];
            u16* dst = &Abuf[row * APITCH + c4 * 4];
            dst[0] = f2bf(v.x); dst[1] = f2bf(v.y);
            dst[2] = f2bf(v.z); dst[3] = f2bf(v.w);
        }
#else
        #pragma unroll
        for (int i = 0; i < 16; ++i) {
            int idx = tid + i * 256;
            int row = idx >> 7;
            int c4  = idx & 127;
            const float4 v =
                reinterpret_cast<const float4*>(xbase + (size_t)row * F_SZ)[c4];
            u16* dst = &Abuf[row * APITCH + c4 * 4];
            dst[0] = f2bf(v.x); dst[1] = f2bf(v.y);
            dst[2] = f2bf(v.z); dst[3] = f2bf(v.w);
        }
#endif
        __syncthreads();

        #pragma unroll
        for (int j = 0; j < 10; ++j) {
            v8f z = {0.f,0.f,0.f,0.f,0.f,0.f,0.f,0.f};
            acc0[j] = z; acc1[j] = z;
        }

        // ---- GEMM: two 16-row tiles per B fragment, F = 16 k-tiles ----
        for (int ft = 0; ft < 16; ++ft) {
            const u16* a0 = &Abuf[m * APITCH + ft * 32 + (hi ? 8 : 0)];
            const u16* a1 = &Abuf[(16 + m) * APITCH + ft * 32 + (hi ? 8 : 0)];
            v16u av0, av1;
            #pragma unroll
            for (int h = 0; h < 8; ++h) {
                av0[h] = a0[h]; av0[8 + h] = a0[16 + h];
                av1[h] = a1[h]; av1[8 + h] = a1[16 + h];
            }
            v16bf aF0 = __builtin_bit_cast(v16bf, av0);
            v16bf aF1 = __builtin_bit_cast(v16bf, av1);

            #pragma unroll
            for (int j = 0; j < 10; ++j) {
                int p  = wave * 10 + j;         // 80 (k,nt) tiles over 8 waves
                int k  = p >> 3;
                int nt = p & 7;
                const u16* bsrc =
                    packedB + ((size_t)(((k * 16 + ft) * 8 + nt) * 32 + lane) << 4);
                if (j == 0 && ft < 15) {
                    int pn = wave * 10;
                    __builtin_prefetch(
                        packedB + ((size_t)((((pn >> 3) * 16 + ft + 1) * 8 + (pn & 7)) * 32 + lane) << 4),
                        0, 1);
                }
                v16u bv;
                #pragma unroll
                for (int h = 0; h < 16; ++h) bv[h] = bsrc[h];
                v16bf bF = __builtin_bit_cast(v16bf, bv);
                acc0[j] = __builtin_amdgcn_wmma_f32_16x16x32_bf16(
                              false, aF0, false, bF, (short)0, acc0[j], false, false);
                acc1[j] = __builtin_amdgcn_wmma_f32_16x16x32_bf16(
                              false, aF1, false, bF, (short)0, acc1[j], false, false);
            }
        }

        // ---- dump D (+bias) into LDS M tile ----
        #pragma unroll
        for (int j = 0; j < 10; ++j) {
            int p  = wave * 10 + j;
            int k  = p >> 3;
            int nt = p & 7;
            int uu = nt * 16 + m;
            float bv = bias[k * U_SZ + uu];
            #pragma unroll
            for (int g = 0; g < 8; ++g) {
                int row = g + (hi << 3);
                Mbuf[(k * ST + row) * UPITCH + uu]        = acc0[j][g] + bv;
                Mbuf[(k * ST + 16 + row) * UPITCH + uu]   = acc1[j][g] + bv;
            }
        }
        __syncthreads();

        // ---- sequential in-tile scan (exclusive-cumsum chains) ----
        #define MV(kk) Mbuf[((kk) * ST + t) * UPITCH + u]
        if (grp == 0) {
            for (int t = 0; t < ST; ++t) {
                y1 += MV(0);                                          // level 1
                float a1 = MV(6), a2 = MV(7), a3 = MV(8), a4 = MV(9); // level 4
                float r12 = a2 * s11, r13 = a3 * s12, r14 = a4 * s13;
                float r23 = a3 * s22, r24 = a4 * s23;
                float r34 = a4 * s33;
                s11 += a1; s12 += r12; s13 += r13; s14 += r14;
                s22 += a2; s23 += r23; s24 += r24;
                s33 += a3; s34 += r34;
                s44 += a4;
            }
        } else {
            for (int t = 0; t < ST; ++t) {
                float b1 = MV(1), b2 = MV(2);                         // level 2
                float r2 = b2 * t11;
                t11 += b1; t12 += r2; t22 += b2;
                float c1 = MV(3), c2 = MV(4), c3 = MV(5);             // level 3
                float ra = c2 * q11, rb = c3 * q12, rc = c3 * q22;
                q11 += c1; q12 += ra; q13 += rb;
                q22 += c2; q23 += rc; q33 += c3;
            }
        }
        #undef MV
        __syncthreads();
    }

    // ---- write per-chunk sub-chain partials ----
    float* pb = partials + (size_t)blockIdx.x * NSLOT * U_SZ + u;
    if (grp == 0) {
        pb[ 0 * U_SZ] = y1;
        pb[10 * U_SZ] = s11; pb[11 * U_SZ] = s12; pb[12 * U_SZ] = s13; pb[13 * U_SZ] = s14;
        pb[14 * U_SZ] = s22; pb[15 * U_SZ] = s23; pb[16 * U_SZ] = s24;
        pb[17 * U_SZ] = s33; pb[18 * U_SZ] = s34; pb[19 * U_SZ] = s44;
    } else {
        pb[ 1 * U_SZ] = t11; pb[ 2 * U_SZ] = t12; pb[ 3 * U_SZ] = t22;
        pb[ 4 * U_SZ] = q11; pb[ 5 * U_SZ] = q12; pb[ 6 * U_SZ] = q13;
        pb[ 7 * U_SZ] = q22; pb[ 8 * U_SZ] = q23; pb[ 9 * U_SZ] = q33;
    }
}

// ---------------------------------------------------------------------------
// Cross-chunk combine: P'_j = sum_{i<=j} P_i * S_{i+1..j}  (P_0 = 1)
// ---------------------------------------------------------------------------
__global__ __launch_bounds__(256)
void ls2t_combine(const float* __restrict__ partials, float* __restrict__ out) {
    int gid = blockIdx.x * 256 + threadIdx.x;   // 32*128 = 4096
    int b = gid >> 7, u = gid & 127;
    float y1 = 0.f;
    float p21=0,p22=0;
    float p31=0,p32=0,p33=0;
    float p41=0,p42=0,p43=0,p44=0;
    for (int c = 0; c < CHUNKS; ++c) {
        const float* S = partials + (size_t)(b * CHUNKS + c) * NSLOT * U_SZ + u;
        #define SL(i) S[(i) * U_SZ]
        y1 += SL(0);
        { float n1 = p21 + SL(1);
          float n2 = p22 + p21 * SL(3) + SL(2);
          p21 = n1; p22 = n2; }
        { float n1 = p31 + SL(4);
          float n2 = p32 + p31 * SL(7) + SL(5);
          float n3 = p33 + p32 * SL(9) + p31 * SL(8) + SL(6);
          p31 = n1; p32 = n2; p33 = n3; }
        { float n1 = p41 + SL(10);
          float n2 = p42 + p41 * SL(14) + SL(11);
          float n3 = p43 + p42 * SL(17) + p41 * SL(15) + SL(12);
          float n4 = p44 + p43 * SL(19) + p42 * SL(18) + p41 * SL(16) + SL(13);
          p41 = n1; p42 = n2; p43 = n3; p44 = n4; }
        #undef SL
    }
    float* o = out + (size_t)b * 4 * U_SZ + u;
    o[0 * U_SZ] = y1;
    o[1 * U_SZ] = p22;
    o[2 * U_SZ] = p33;
    o[3 * U_SZ] = p44;
}

extern "C" void kernel_launch(void* const* d_in, const int* in_sizes, int n_in,
                              void* d_out, int out_size, void* d_ws, size_t ws_size,
                              hipStream_t stream) {
    (void)in_sizes; (void)n_in; (void)out_size; (void)ws_size;
    const float* X    = (const float*)d_in[0];   // [32,2048,512]
    const float* kern = (const float*)d_in[1];   // [10,512,128]
    const float* bias = (const float*)d_in[2];   // [10,128]
    float* out = (float*)d_out;                  // [32,4,128]

    u16*   packedB  = (u16*)d_ws;                                 // 1.25 MB
    float* partials = (float*)((char*)d_ws + PACKED_BYTES);       // 5.0 MB

    pack_weights<<<(PACKED_ELEMS + 255) / 256, 256, 0, stream>>>(kern, packedB);
    ls2t_main<<<B_SZ * CHUNKS, 256, 0, stream>>>(X, bias, packedB, partials);
    ls2t_combine<<<(B_SZ * U_SZ) / 256, 256, 0, stream>>>(partials, out);
}
